// CluttrVAE_39290360823849
// MI455X (gfx1250) — matrix operand order repaired and checked by reading, not compiled, not run on hardware
//
#include <hip/hip_runtime.h>
#include <math.h>

// ---------------------------------------------------------------------------
// CDNA5 (gfx1250) WMMA bf16 implementation of the CluttrVAE forward pass.
// wave32, V_WMMA_F32_16X16X32_BF16, LDS double-buffered GEMM tiles with
// bank-conflict-free padding and (when available) async global->LDS staging.
// ---------------------------------------------------------------------------

typedef __attribute__((ext_vector_type(16))) __bf16 v16bf;
typedef __attribute__((ext_vector_type(8)))  float  v8f;

#if defined(__has_builtin)
#if __has_builtin(__builtin_amdgcn_global_load_async_to_lds_b128) && \
    __has_builtin(__builtin_amdgcn_s_wait_asynccnt)
#define HAVE_ASYNC_LDS 1
#endif
#endif
#ifndef HAVE_ASYNC_LDS
#define HAVE_ASYNC_LDS 0
#endif

#if HAVE_ASYNC_LDS
// Builtin expects pointers to 16-byte int vectors in AS1 (global) / AS3 (LDS).
typedef int v4i_vec __attribute__((vector_size(16)));
typedef __attribute__((address_space(1))) v4i_vec v4i_glob;
typedef __attribute__((address_space(3))) v4i_vec v4i_lds;
__device__ __forceinline__ void async_cp16(const unsigned short* g, unsigned short* l) {
  // 16-byte global -> LDS DMA copy, tracked by ASYNCcnt.
  __builtin_amdgcn_global_load_async_to_lds_b128((v4i_glob*)g, (v4i_lds*)l, 0, 0);
}
__device__ __forceinline__ void async_fence() {
  __builtin_amdgcn_s_wait_asynccnt(0);
}
#endif

// LDS row stride for 32-K bf16 tiles: 40 elements (80B = 20 banks).
// 20*r mod 64 is distinct for r = 0..15 -> conflict-free ds_load_b128
// across the 16 row-lanes of a WMMA fragment read.
#define LDS_K 40

struct Frag32B { uint4 lo; uint4 hi; };

__device__ __forceinline__ v16bf frag_cast(uint4 lo, uint4 hi) {
  Frag32B s{lo, hi};
  return __builtin_bit_cast(v16bf, s);
}

__device__ __forceinline__ unsigned short f2bf(float f) {
  unsigned int u = __float_as_uint(f);
  u += 0x7FFFu + ((u >> 16) & 1u);          // round-to-nearest-even
  return (unsigned short)(u >> 16);
}

// A-operand fragment (16x32 bf16, M-major rows of stride ld):
// lane-half h holds K = h*8 .. h*8+7  and  h*8+16 .. h*8+23 (ISA 7.12.2)
__device__ __forceinline__ v16bf fragA16(const unsigned short* base, int ld, int row0) {
  int lane = threadIdx.x & 31;
  const unsigned short* p = base + (size_t)(row0 + (lane & 15)) * ld + (lane >> 4) * 8;
  return frag_cast(*(const uint4*)p, *(const uint4*)(p + 16));
}

// B-operand fragment (32x16 bf16 fed from N-major "Bt" rows of stride ld):
// lane-half h holds K = h*8 .. h*8+7  and  h*8+16 .. h*8+23 (matches A)
__device__ __forceinline__ v16bf fragB16(const unsigned short* base, int ld, int row0) {
  int lane = threadIdx.x & 31;
  const unsigned short* p = base + (size_t)(row0 + (lane & 15)) * ld + (lane >> 4) * 8;
  return frag_cast(*(const uint4*)p, *(const uint4*)(p + 16));
}

__device__ __forceinline__ v8f wmma_bf16(v16bf a, v16bf b, v8f c) {
  // (neg_a, A, neg_b, B, c_mod, C, reuse_a, reuse_b)
  return __builtin_amdgcn_wmma_f32_16x16x32_bf16(false, a, false, b, (short)0, c,
                                                 false, false);
}

__device__ __forceinline__ float act_apply(float v, int act) {
  if (act == 1) return v > 0.0f ? v : 0.0f;                 // relu
  if (act == 2) return 1.0f / (1.0f + __expf(-v));          // sigmoid
  return v;
}

// ---------------------------------------------------------------------------
// Generic GEMM:  C[M,N] = act(A[M,Kp] * Bt[N,Kp]^T + bias)
// A row-major bf16 (lda = Kp), Bt is the N-major bf16 transposed weight.
// Output: optional f32 (ldcf) and/or bf16 (ldcb; pad cols [N,ldcb) zeroed).
// Tile 128x128x32, 8 waves, double-buffered LDS, async staging on interior
// tiles when the toolchain exposes the gfx1250 async-LDS builtins.
// ---------------------------------------------------------------------------
__global__ __launch_bounds__(256) void gemm_bf16_wmma(
    const unsigned short* __restrict__ A, int lda,
    const unsigned short* __restrict__ Bt, int ldb,
    const float* __restrict__ bias, int act,
    float* __restrict__ Cf, int ldcf,
    unsigned short* __restrict__ Cb, int ldcb,
    int M, int N, int Kp)
{
  __shared__ unsigned short sA[2][128 * LDS_K];
  __shared__ unsigned short sB[2][128 * LDS_K];

  const int tid  = threadIdx.x;
  const int lane = tid & 31;
  const int wv   = tid >> 5;
  const int half = lane >> 4;
  const int mBase = blockIdx.y * 128;
  const int nBase = blockIdx.x * 128;
  const int m0 = (wv & 3) * 32;   // wave covers 32 rows
  const int n0 = (wv >> 2) * 64;  // wave covers 64 cols
  const bool interior = (mBase + 128 <= M) && (nBase + 128 <= N);

  const v8f vz = {0.f, 0.f, 0.f, 0.f, 0.f, 0.f, 0.f, 0.f};
  v8f acc[2][4];
  for (int i = 0; i < 2; ++i)
    for (int j = 0; j < 4; ++j) acc[i][j] = vz;

  // Each thread stages 2x16B for A and 2x16B for B per K-tile.
  const int sr  = tid >> 2;          // row 0..63 (+64 on second pass)
  const int skc = (tid & 3) * 8;     // k-chunk 0/8/16/24

  auto stage_fast = [&](int kb, int buf) {
#pragma unroll
    for (int i = 0; i < 2; ++i) {
      int r = sr + i * 64;
      const unsigned short* ga = A + (size_t)(mBase + r) * lda + kb + skc;
      const unsigned short* gb = Bt + (size_t)(nBase + r) * ldb + kb + skc;
      unsigned short* la = &sA[buf][r * LDS_K + skc];
      unsigned short* lb = &sB[buf][r * LDS_K + skc];
#if HAVE_ASYNC_LDS
      async_cp16(ga, la);
      async_cp16(gb, lb);
#else
      *(uint4*)la = *(const uint4*)ga;
      *(uint4*)lb = *(const uint4*)gb;
#endif
      if (kb + 64 < Kp) {  // prefetch K+2 tile -> global_prefetch_b8
        __builtin_prefetch(ga + 64);
        __builtin_prefetch(gb + 64);
      }
    }
  };

  auto stage_edge = [&](int kb, int buf) {
#pragma unroll
    for (int i = 0; i < 2; ++i) {
      int r = sr + i * 64;
      uint4 va = make_uint4(0, 0, 0, 0);
      uint4 vb = make_uint4(0, 0, 0, 0);
      if (mBase + r < M) va = *(const uint4*)(A + (size_t)(mBase + r) * lda + kb + skc);
      if (nBase + r < N) vb = *(const uint4*)(Bt + (size_t)(nBase + r) * ldb + kb + skc);
      *(uint4*)(&sA[buf][r * LDS_K + skc]) = va;
      *(uint4*)(&sB[buf][r * LDS_K + skc]) = vb;
    }
  };

  const int nk = Kp >> 5;
  if (interior) stage_fast(0, 0); else stage_edge(0, 0);
#if HAVE_ASYNC_LDS
  if (interior) async_fence();
#endif
  __syncthreads();
  for (int tk = 0; tk < nk; ++tk) {
    int buf = tk & 1;
    if (tk + 1 < nk) {
      if (interior) stage_fast((tk + 1) << 5, buf ^ 1);
      else          stage_edge((tk + 1) << 5, buf ^ 1);
    }
    v16bf af0 = fragA16(sA[buf], LDS_K, m0);
    v16bf af1 = fragA16(sA[buf], LDS_K, m0 + 16);
    v16bf bf0 = fragB16(sB[buf], LDS_K, n0);
    v16bf bf1 = fragB16(sB[buf], LDS_K, n0 + 16);
    v16bf bf2 = fragB16(sB[buf], LDS_K, n0 + 32);
    v16bf bf3 = fragB16(sB[buf], LDS_K, n0 + 48);
    acc[0][0] = wmma_bf16(af0, bf0, acc[0][0]);
    acc[0][1] = wmma_bf16(af0, bf1, acc[0][1]);
    acc[0][2] = wmma_bf16(af0, bf2, acc[0][2]);
    acc[0][3] = wmma_bf16(af0, bf3, acc[0][3]);
    acc[1][0] = wmma_bf16(af1, bf0, acc[1][0]);
    acc[1][1] = wmma_bf16(af1, bf1, acc[1][1]);
    acc[1][2] = wmma_bf16(af1, bf2, acc[1][2]);
    acc[1][3] = wmma_bf16(af1, bf3, acc[1][3]);
#if HAVE_ASYNC_LDS
    if (interior) async_fence();   // wave-local wait for tile k+1 DMA
#endif
    __syncthreads();
  }

  // C layout (ISA 7.12.2): VGPR r -> M = r + half*8, N = lane&15
  for (int i = 0; i < 2; ++i) {
    for (int j = 0; j < 4; ++j) {
      int col = nBase + n0 + j * 16 + (lane & 15);
      int rB  = mBase + m0 + i * 16 + half * 8;
#pragma unroll
      for (int r = 0; r < 8; ++r) {
        int row = rB + r;
        if (row >= M) continue;
        if (col < N) {
          float v = acc[i][j][r];
          if (bias) v += bias[col];
          v = act_apply(v, act);
          if (Cf) Cf[(size_t)row * ldcf + col] = v;
          if (Cb) Cb[(size_t)row * ldcb + col] = f2bf(v);
        } else if (Cb && col < ldcb) {
          Cb[(size_t)row * ldcb + col] = 0;  // zero K-padding of bf16 activations
        }
      }
    }
  }
}

// ---------------------------------------------------------------------------
// Fused LSTM recurrent-gate GEMM, one time step, both directions (gridDim.y=2):
//   gates[dir][32,4H] = xg[dir][row(b,tt)] + h_bf16[dir][32,Kp] @ WhT[dir][4H,Kp]^T
// Wave handles a 16-wide N slice with two 16-row A fragments (M = B = 32).
// Operands are L2-resident; fragments load straight from global.
// ---------------------------------------------------------------------------
__global__ __launch_bounds__(256) void lstm_gates_wmma(
    const unsigned short* __restrict__ hbf, int ldh,
    const unsigned short* __restrict__ Wh0,
    const unsigned short* __restrict__ Wh1, int ldw,
    const float* __restrict__ xg0, const float* __restrict__ xg1,
    long sB, long sT,
    float* __restrict__ gates, int fourH, int Kp, int t, int S)
{
  const int dir  = blockIdx.y;
  const int tid  = threadIdx.x;
  const int lane = tid & 31;
  const int wv   = tid >> 5;
  const int n0   = (blockIdx.x * 8 + wv) * 16;
  if (n0 >= fourH) return;  // wave-uniform branch, EXEC stays all-ones inside

  const unsigned short* Ah = hbf + (size_t)dir * 32 * ldh;
  const unsigned short* Bt = dir ? Wh1 : Wh0;
  const float* xg = dir ? xg1 : xg0;
  const int tt = dir ? (S - 1 - t) : t;

  const v8f vz = {0.f, 0.f, 0.f, 0.f, 0.f, 0.f, 0.f, 0.f};
  v8f acc0 = vz, acc1 = vz;
  for (int kb = 0; kb < Kp; kb += 32) {
    v16bf a0 = fragA16(Ah + kb, ldh, 0);
    v16bf a1 = fragA16(Ah + kb, ldh, 16);
    v16bf b0 = fragB16(Bt + kb, ldw, n0);
    acc0 = wmma_bf16(a0, b0, acc0);
    acc1 = wmma_bf16(a1, b0, acc1);
  }
  const int half = lane >> 4;
  const int col  = n0 + (lane & 15);
  float* gout = gates + (size_t)dir * 32 * fourH;
#pragma unroll
  for (int r = 0; r < 8; ++r) {
    int b0r = r + half * 8;
    int b1r = b0r + 16;
    gout[(size_t)b0r * fourH + col] = acc0[r] + xg[(size_t)b0r * sB + (size_t)tt * sT + col];
    gout[(size_t)b1r * fourH + col] = acc1[r] + xg[(size_t)b1r * sB + (size_t)tt * sT + col];
  }
}

// LSTM cell pointwise update; optionally scatters h into the output sequence.
__global__ void lstm_pointwise(const float* __restrict__ gates, int fourH,
                               float* __restrict__ cst,
                               unsigned short* __restrict__ hbf, int ldh,
                               unsigned short* __restrict__ outSeq, int ldo,
                               int H, int t, int S)
{
  int dir = blockIdx.y;
  int i = blockIdx.x * blockDim.x + threadIdx.x;
  if (i >= 32 * H) return;
  int b = i / H, n = i - b * H;
  const float* g = gates + ((size_t)dir * 32 + b) * fourH;
  float gi = 1.0f / (1.0f + __expf(-g[n]));
  float gf = 1.0f / (1.0f + __expf(-g[n + H]));
  float gg = tanhf(g[n + 2 * H]);
  float go = 1.0f / (1.0f + __expf(-g[n + 3 * H]));
  float* cp = cst + ((size_t)dir * 32 + b) * H + n;
  float c = gf * (*cp) + gi * gg;
  *cp = c;
  float h = go * tanhf(c);
  hbf[((size_t)dir * 32 + b) * ldh + n] = f2bf(h);
  if (outSeq) {
    int tt = dir ? (S - 1 - t) : t;
    outSeq[((size_t)b * S + tt) * ldo + dir * H + n] = f2bf(h);
  }
}

// W[K][N] f32 -> Wt[N][Kp] bf16 (zero K-padding), LDS-tiled transpose.
__global__ void transpose_cast_f32_bf16(const float* __restrict__ W, int K, int N,
                                        unsigned short* __restrict__ Wt, int Kp)
{
  __shared__ float tile[32][33];
  int nb = blockIdx.x * 32;
  int kb = blockIdx.y * 32;
#pragma unroll
  for (int j = 0; j < 4; ++j) {
    int k = kb + threadIdx.y + j * 8;
    int n = nb + threadIdx.x;
    tile[threadIdx.y + j * 8][threadIdx.x] = (k < K && n < N) ? W[(size_t)k * N + n] : 0.0f;
  }
  __syncthreads();
#pragma unroll
  for (int j = 0; j < 4; ++j) {
    int n = nb + threadIdx.y + j * 8;
    int k = kb + threadIdx.x;
    if (n < N && k < Kp) Wt[(size_t)n * Kp + k] = f2bf(tile[threadIdx.x][threadIdx.y + j * 8]);
  }
}

__global__ void embed_gather(const float* __restrict__ embed, const int* __restrict__ x,
                             unsigned short* __restrict__ out, int E, int EP, int rows)
{
  int i = blockIdx.x * blockDim.x + threadIdx.x;
  if (i >= rows * EP) return;
  int r = i / EP, e = i - r * EP;
  out[i] = (e < E) ? f2bf(embed[(size_t)x[r] * E + e]) : (unsigned short)0;
}

__global__ void highway_combine(const float* __restrict__ fg, const float* __restrict__ q,
                                const float* __restrict__ gate,
                                unsigned short* __restrict__ out, int E, int EP, int rows)
{
  int i = blockIdx.x * blockDim.x + threadIdx.x;
  if (i >= rows * EP) return;
  int r = i / EP, e = i - r * EP;
  if (e < E) {
    float gt = gate[(size_t)r * E + e];
    out[i] = f2bf(gt * fg[(size_t)r * E + e] + (1.0f - gt) * q[(size_t)r * E + e]);
  } else {
    out[i] = 0;
  }
}

// enc_h[b] = concat(h_fwd_final[b], h_bwd_final[b]), bf16, padded to H2EP.
__global__ void build_ench(const unsigned short* __restrict__ hbf, int ldh,
                           unsigned short* __restrict__ out, int HE, int H2EP)
{
  int i = blockIdx.x * blockDim.x + threadIdx.x;
  if (i >= 32 * H2EP) return;
  int b = i / H2EP, k = i - b * H2EP;
  unsigned short v = 0;
  if (k < HE)            v = hbf[(size_t)b * ldh + k];
  else if (k < 2 * HE)   v = hbf[(size_t)(32 + b) * ldh + (k - HE)];
  out[i] = v;
}

// z = mean + eps * exp(0.5*logvar) -> bf16
__global__ void z_make(const float* __restrict__ mean, const float* __restrict__ logvar,
                       const float* __restrict__ eps, unsigned short* __restrict__ zb, int n)
{
  int i = blockIdx.x * blockDim.x + threadIdx.x;
  if (i < n) zb[i] = f2bf(mean[i] + eps[i] * __expf(0.5f * logvar[i]));
}

// ---------------------------------------------------------------------------
// Host orchestration
// ---------------------------------------------------------------------------
extern "C" void kernel_launch(void* const* d_in, const int* in_sizes, int n_in,
                              void* d_out, int out_size, void* d_ws, size_t ws_size,
                              hipStream_t stream)
{
  // Model dims
  const int Bb = 32, S = 128, V = 20000, E = 300, L = 128;
  const int HEc = 300, HDc = 400;
  const int M = Bb * S;                   // 4096 rows
  const int EP = 320;                     // E padded to 32
  const int HEP = 320;                    // H_ENC padded
  const int H2EP = 608;                   // 2*H_ENC padded
  const int HDP = 416;                    // H_DEC padded
  const int H2D = 800;                    // 2*H_DEC (already /32)
  const int FHE = 4 * HEc;                // 1200
  const int FHD = 4 * HDc;                // 1600

  // Input tensor indices (setup_inputs dict insertion order, nested dicts
  // flattened recursively in insertion order; dense = {W,b}, lstm = {Wi,Wh,b}):
  // 0:x 1:eps 2:embed 3..12:hw1 13..22:hw2 23..25:enc_f 26..28:enc_b
  // 29,30:mean 31,32:logvar 33..35:dec1_f 36..38:dec1_b 39..41:dec2_f
  // 42..44:dec2_b 45,46:out
  const int* x = (const int*)d_in[0];
  const float* eps = (const float*)d_in[1];
  auto F = [&](int i) { return (const float*)d_in[i]; };

  // workspace allocator (deterministic layout per call)
  size_t off = 0;
  auto alloc = [&](size_t bytes) -> void* {
    off = (off + 255) & ~(size_t)255;
    void* p = (char*)d_ws + off;
    off += bytes;
    return p;
  };

  auto tcast = [&](const float* W, int K, int N, int Kp) -> unsigned short* {
    unsigned short* Wt = (unsigned short*)alloc((size_t)N * Kp * 2);
    dim3 g((N + 31) / 32, (Kp + 31) / 32), b(32, 8);
    transpose_cast_f32_bf16<<<g, b, 0, stream>>>(W, K, N, Wt, Kp);
    return Wt;
  };

  auto gemm = [&](const unsigned short* A, int lda, const unsigned short* Bt, int ldb,
                  const float* bias, int act, float* Cf, int ldcf,
                  unsigned short* Cb, int ldcb, int Mm, int Nn, int Kp) {
    dim3 g((Nn + 127) / 128, (Mm + 127) / 128);
    gemm_bf16_wmma<<<g, 256, 0, stream>>>(A, lda, Bt, ldb, bias, act,
                                          Cf, ldcf, Cb, ldcb, Mm, Nn, Kp);
  };

  // ---- weight transposition + bf16 cast ----
  const unsigned short* hwT[2][5];
  const float* hwB[2][5];
  for (int l = 0; l < 2; ++l)
    for (int m = 0; m < 5; ++m) {
      int idx = 3 + l * 10 + m * 2;
      hwT[l][m] = tcast(F(idx), E, E, EP);
      hwB[l][m] = F(idx + 1);
    }
  const unsigned short *encWiT[2], *encWhT[2];
  const float* encB[2];
  for (int d = 0; d < 2; ++d) {
    encWiT[d] = tcast(F(23 + d * 3), E, FHE, EP);
    encWhT[d] = tcast(F(24 + d * 3), HEc, FHE, HEP);
    encB[d]   = F(25 + d * 3);
  }
  const unsigned short* meanT = tcast(F(29), 2 * HEc, L, H2EP);
  const float* meanB = F(30);
  const unsigned short* logvT = tcast(F(31), 2 * HEc, L, H2EP);
  const float* logvB = F(32);
  const unsigned short *d1WiT[2], *d1WhT[2], *d2WiT[2], *d2WhT[2];
  const float *d1B[2], *d2B[2];
  for (int d = 0; d < 2; ++d) {
    d1WiT[d] = tcast(F(33 + d * 3), L, FHD, L);
    d1WhT[d] = tcast(F(34 + d * 3), HDc, FHD, HDP);
    d1B[d]   = F(35 + d * 3);
    d2WiT[d] = tcast(F(39 + d * 3), H2D, FHD, H2D);
    d2WhT[d] = tcast(F(40 + d * 3), HDc, FHD, HDP);
    d2B[d]   = F(41 + d * 3);
  }
  const unsigned short* outT = tcast(F(45), H2D, V, H2D);
  const float* outB = F(46);

  // ---- activation buffers ----
  unsigned short* embA = (unsigned short*)alloc((size_t)M * EP * 2);
  unsigned short* tmp1 = (unsigned short*)alloc((size_t)M * EP * 2);
  float* fgF   = (float*)alloc((size_t)M * E * 4);
  float* qF    = (float*)alloc((size_t)M * E * 4);
  float* gateF = (float*)alloc((size_t)M * E * 4);
  unsigned short* hxA = (unsigned short*)alloc((size_t)M * EP * 2);
  unsigned short* hxB = (unsigned short*)alloc((size_t)M * EP * 2);
  float* xgE[2]; xgE[0] = (float*)alloc((size_t)M * FHE * 4);
                 xgE[1] = (float*)alloc((size_t)M * FHE * 4);
  float* gatesB = (float*)alloc((size_t)2 * 32 * FHD * 4);
  unsigned short* hbf = (unsigned short*)alloc((size_t)2 * 32 * HDP * 2);
  float* cst = (float*)alloc((size_t)2 * 32 * HDc * 4);
  unsigned short* ench = (unsigned short*)alloc((size_t)32 * H2EP * 2);
  unsigned short* zb   = (unsigned short*)alloc((size_t)32 * L * 2);
  float* xgD1[2]; xgD1[0] = (float*)alloc((size_t)32 * FHD * 4);
                  xgD1[1] = (float*)alloc((size_t)32 * FHD * 4);
  unsigned short* d1 = (unsigned short*)alloc((size_t)M * H2D * 2);
  float* xgD2[2]; xgD2[0] = (float*)alloc((size_t)M * FHD * 4);
                  xgD2[1] = (float*)alloc((size_t)M * FHD * 4);
  unsigned short* d2 = (unsigned short*)alloc((size_t)M * H2D * 2);
  (void)ws_size; (void)in_sizes; (void)n_in; (void)out_size;

  float* outLogits = (float*)d_out;
  float* outMean   = outLogits + (size_t)M * V;
  float* outLogvar = outMean + (size_t)Bb * L;

  const int PT = 256;

  // ---- embedding gather (fp32 -> bf16, K-padded) ----
  embed_gather<<<(M * EP + PT - 1) / PT, PT, 0, stream>>>(F(2), x, embA, E, EP, M);

  // ---- two highway layers ----
  auto highway = [&](const unsigned short* X, unsigned short* out, int l) {
    gemm(X, EP, hwT[l][0], EP, hwB[l][0], 1, nullptr, 0, tmp1, EP, M, E, EP);  // g = relu
    gemm(tmp1, EP, hwT[l][1], EP, hwB[l][1], 1, fgF, E, nullptr, 0, M, E, EP); // f_g = relu
    gemm(X, EP, hwT[l][2], EP, hwB[l][2], 1, nullptr, 0, tmp1, EP, M, E, EP);  // q1 = relu
    gemm(tmp1, EP, hwT[l][3], EP, hwB[l][3], 0, qF, E, nullptr, 0, M, E, EP);  // q
    gemm(X, EP, hwT[l][4], EP, hwB[l][4], 2, gateF, E, nullptr, 0, M, E, EP);  // gate = sig
    highway_combine<<<(M * EP + PT - 1) / PT, PT, 0, stream>>>(fgF, qF, gateF, out, E, EP, M);
  };
  highway(embA, hxA, 0);
  highway(hxA, hxB, 1);

  // ---- encoder bi-LSTM (only final h per direction needed) ----
  for (int d = 0; d < 2; ++d)
    gemm(hxB, EP, encWiT[d], EP, encB[d], 0, xgE[d], FHE, nullptr, 0, M, FHE, EP);
  (void)hipMemsetAsync(hbf, 0, (size_t)2 * 32 * HDP * 2, stream);
  (void)hipMemsetAsync(cst, 0, (size_t)2 * 32 * HDc * 4, stream);
  for (int t = 0; t < S; ++t) {
    lstm_gates_wmma<<<dim3((FHE + 127) / 128, 2), 256, 0, stream>>>(
        hbf, HDP, encWhT[0], encWhT[1], HEP, xgE[0], xgE[1],
        (long)S * FHE, (long)FHE, gatesB, FHE, HEP, t, S);
    lstm_pointwise<<<dim3((32 * HEc + PT - 1) / PT, 2), PT, 0, stream>>>(
        gatesB, FHE, cst, hbf, HDP, nullptr, 0, HEc, t, S);
  }
  build_ench<<<(32 * H2EP + PT - 1) / PT, PT, 0, stream>>>(hbf, HDP, ench, HEc, H2EP);

  // ---- mean / logvar / z ----
  gemm(ench, H2EP, meanT, H2EP, meanB, 0, outMean, L, nullptr, 0, Bb, L, H2EP);
  gemm(ench, H2EP, logvT, H2EP, logvB, 0, outLogvar, L, nullptr, 0, Bb, L, H2EP);
  z_make<<<(Bb * L + PT - 1) / PT, PT, 0, stream>>>(outMean, outLogvar, eps, zb, Bb * L);

  // ---- decoder bi-LSTM 1 (input z is time-invariant -> xg has no time dim) ----
  for (int d = 0; d < 2; ++d)
    gemm(zb, L, d1WiT[d], L, d1B[d], 0, xgD1[d], FHD, nullptr, 0, Bb, FHD, L);
  (void)hipMemsetAsync(hbf, 0, (size_t)2 * 32 * HDP * 2, stream);
  (void)hipMemsetAsync(cst, 0, (size_t)2 * 32 * HDc * 4, stream);
  for (int t = 0; t < S; ++t) {
    lstm_gates_wmma<<<dim3((FHD + 127) / 128, 2), 256, 0, stream>>>(
        hbf, HDP, d1WhT[0], d1WhT[1], HDP, xgD1[0], xgD1[1],
        (long)FHD, 0L, gatesB, FHD, HDP, t, S);
    lstm_pointwise<<<dim3((32 * HDc + PT - 1) / PT, 2), PT, 0, stream>>>(
        gatesB, FHD, cst, hbf, HDP, d1, H2D, HDc, t, S);
  }

  // ---- decoder bi-LSTM 2 ----
  for (int d = 0; d < 2; ++d)
    gemm(d1, H2D, d2WiT[d], H2D, d2B[d], 0, xgD2[d], FHD, nullptr, 0, M, FHD, H2D);
  (void)hipMemsetAsync(hbf, 0, (size_t)2 * 32 * HDP * 2, stream);
  (void)hipMemsetAsync(cst, 0, (size_t)2 * 32 * HDc * 4, stream);
  for (int t = 0; t < S; ++t) {
    lstm_gates_wmma<<<dim3((FHD + 127) / 128, 2), 256, 0, stream>>>(
        hbf, HDP, d2WhT[0], d2WhT[1], HDP, xgD2[0], xgD2[1],
        (long)S * FHD, (long)FHD, gatesB, FHD, HDP, t, S);
    lstm_pointwise<<<dim3((32 * HDc + PT - 1) / PT, 2), PT, 0, stream>>>(
        gatesB, FHD, cst, hbf, HDP, d2, H2D, HDc, t, S);
  }

  // ---- output projection: [4096,800] x [800,20000] -> fp32 logits ----
  gemm(d2, H2D, outT, H2D, outB, 0, outLogits, V, nullptr, 0, M, V, H2D);
}
